// HyperLSTMCell_74783970558044
// MI455X (gfx1250) — matrix operand order repaired and checked by reading, not compiled, not run on hardware
//
#include <hip/hip_runtime.h>
#include <hip/hip_bf16.h>

typedef __attribute__((ext_vector_type(16))) _Float16 v16h;
typedef __attribute__((ext_vector_type(8)))  _Float16 v8h;
typedef __attribute__((ext_vector_type(8)))  float    v8f;

namespace {
constexpr int kB  = 4096;   // batch
constexpr int kD  = 512;    // input dim
constexpr int kH  = 1024;   // main hidden
constexpr int kHH = 256;    // hyper hidden
constexpr int kE  = 32;     // embedding
constexpr float kEps = 1e-5f;
constexpr float kForgetBias = 1.0f;
}

#define DEVFN __device__ __forceinline__

DEVFN float sigmoidf_(float x) { return 1.0f / (1.0f + expf(-x)); }

// Block-wide sum over 256 threads; result broadcast to all threads.
DEVFN float block_sum_256(float x, float* red) {
  int t = threadIdx.x;
  red[t] = x;
  __syncthreads();
  #pragma unroll
  for (int off = 128; off > 0; off >>= 1) {
    if (t < off) red[t] += red[t + off];
    __syncthreads();
  }
  float r = red[0];
  __syncthreads();
  return r;
}

// ---------------------------------------------------------------- conversions
__global__ void cvt_f32_to_f16(const float* __restrict__ src,
                               _Float16* __restrict__ dst, long long n) {
  long long i = (long long)blockIdx.x * blockDim.x + threadIdx.x;
  if (i < n) dst[i] = (_Float16)src[i];
}

// h_total [B x (H+HH)] -> h16 [B x H], hh16 [B x HH]
__global__ __launch_bounds__(256)
void split_h_kernel(const float* __restrict__ h_total,
                    _Float16* __restrict__ h16,
                    _Float16* __restrict__ hh16) {
  int b = blockIdx.x, t = threadIdx.x;
  const float* row = h_total + (size_t)b * (kH + kHH);
  #pragma unroll
  for (int j = 0; j < 4; ++j)
    h16[(size_t)b * kH + j * 256 + t] = (_Float16)row[j * 256 + t];
  hh16[(size_t)b * kHH + t] = (_Float16)row[kH + t];
}

// ---------------------------------------------------------------- WMMA GEMM
// C[M x N] = A[M x LDA] @ W[N x LDW]^T  (f16 in, f32 out), K contraction.
// Compile-time LDA/LDW/LDC/K so load/store addressing folds into immediate
// offsets (24-bit IOFFSET) instead of v_add_nc_u64 chains.
// Block: 128 threads (4 waves). Block tile 64(M) x 128(N); wave tile 32x64.
// M % 64 == 0, N % 128 == 0, K % 32 == 0, LDA/LDW multiples of 8.
template <int LDA, int LDW, int LDC, int K>
__global__ __launch_bounds__(128)
void wmma_gemm_f16(const _Float16* __restrict__ A,
                   const _Float16* __restrict__ W,
                   float* __restrict__ C) {
  const int lane = threadIdx.x & 31;
  const int wave = threadIdx.x >> 5;        // 0..3
  const int r    = lane & 15;
  const int hi   = lane >> 4;               // 0/1
  const int m_blk = blockIdx.y * 64 + (wave >> 1) * 32;
  const int n_blk = blockIdx.x * 128 + (wave & 1) * 64;

  v8f acc[2][4];
  #pragma unroll
  for (int mi = 0; mi < 2; ++mi)
    #pragma unroll
    for (int ni = 0; ni < 4; ++ni)
      #pragma unroll
      for (int j = 0; j < 8; ++j) acc[mi][ni][j] = 0.0f;

  // Per-lane base pointers (row fixed for whole K loop).
  const _Float16* pa = A + (size_t)(m_blk + r) * LDA + hi * 8;
  const _Float16* pw = W + (size_t)(n_blk + r) * LDW + hi * 8;

  #pragma unroll (K == 32 ? 1 : 2)
  for (int k0 = 0; k0 < K; k0 += 32) {
    // A fragments: 16x32, lane r = row, hi selects K base +8.
    // elems 0..7 <- K k0+hi*8+[0..7]; elems 8..15 <- +16.
    v16h a[2];
    #pragma unroll
    for (int mi = 0; mi < 2; ++mi) {
      const _Float16* p = pa + (size_t)(mi * 16) * LDA + k0;
      v8h lo  = *(const v8h*)(p);
      v8h hi8 = *(const v8h*)(p + 16);
      #pragma unroll
      for (int t = 0; t < 8; ++t) { a[mi][t] = lo[t]; a[mi][t + 8] = hi8[t]; }
    }
    // B fragments (= W^T tile): 32x16, lane r = col N, hi selects K base +8.
    // elems 0..7 <- K k0+hi*8+[0..7]; elems 8..15 <- +16.
    v16h bf[4];
    #pragma unroll
    for (int ni = 0; ni < 4; ++ni) {
      const _Float16* p = pw + (size_t)(ni * 16) * LDW + k0;
      v8h lo  = *(const v8h*)(p);
      v8h hi8 = *(const v8h*)(p + 16);
      #pragma unroll
      for (int t = 0; t < 8; ++t) { bf[ni][t] = lo[t]; bf[ni][t + 8] = hi8[t]; }
    }
    #pragma unroll
    for (int mi = 0; mi < 2; ++mi)
      #pragma unroll
      for (int ni = 0; ni < 4; ++ni)
        acc[mi][ni] = __builtin_amdgcn_wmma_f32_16x16x32_f16(
            false, a[mi], false, bf[ni], (short)0, acc[mi][ni], false, false);
  }

  // C/D layout: lane r = N col; VGPR j = M row j + hi*8.
  float* cbase = C + (size_t)(m_blk + hi * 8) * LDC + n_blk + r;
  #pragma unroll
  for (int mi = 0; mi < 2; ++mi)
    #pragma unroll
    for (int ni = 0; ni < 4; ++ni)
      #pragma unroll
      for (int j = 0; j < 8; ++j)
        cbase[(size_t)(mi * 16 + j) * LDC + ni * 16] = acc[mi][ni][j];
}

// ---------------------------------------------------------------- hyper cell
__global__ __launch_bounds__(256)
void hyper_cell_kernel(const float* __restrict__ hyp_wx,
                       const float* __restrict__ hyp_wh,
                       const float* __restrict__ c_total_in,
                       const float* __restrict__ lnh_w, const float* __restrict__ lnh_b,
                       const float* __restrict__ lnc_w, const float* __restrict__ lnc_b,
                       float* __restrict__ hh_new,
                       float* __restrict__ out_h_total,
                       float* __restrict__ out_c_total) {
  __shared__ float red[256];
  int b = blockIdx.x, t = threadIdx.x;
  const float* wx = hyp_wx + (size_t)b * (4 * kHH);
  const float* wh = hyp_wh + (size_t)b * (4 * kHH);
  float v[4], ln[4];
  #pragma unroll
  for (int c = 0; c < 4; ++c) v[c] = wx[c * 256 + t] + wh[c * 256 + t];
  for (int c = 0; c < 4; ++c) {                 // chunk-LN, 4 chunks of 256
    float mu = block_sum_256(v[c], red) * (1.0f / 256.0f);
    float d  = v[c] - mu;
    float var = block_sum_256(d * d, red) * (1.0f / 256.0f);
    ln[c] = d * rsqrtf(var + kEps) * lnh_w[c * 256 + t] + lnh_b[c * 256 + t];
  }
  float c_old = c_total_in[(size_t)b * (kH + kHH) + kH + t];
  float c_new = sigmoidf_(ln[1] + kForgetBias) * c_old +
                sigmoidf_(ln[0]) * tanhf(ln[3]);
  float mu = block_sum_256(c_new, red) * (1.0f / 256.0f);
  float d  = c_new - mu;
  float var = block_sum_256(d * d, red) * (1.0f / 256.0f);
  float lnc = d * rsqrtf(var + kEps) * lnc_w[t] + lnc_b[t];
  float h_new = sigmoidf_(ln[2]) * tanhf(lnc);
  hh_new[(size_t)b * kHH + t] = h_new;
  out_h_total[(size_t)b * (kH + kHH) + kH + t] = h_new;
  out_c_total[(size_t)b * (kH + kHH) + kH + t] = c_new;
}

// -------------------------------------------- first-stage hypernet einsums
// tmp[b, g*32+e] = dot(h_hyper[b,:], w0[g,e,:]) (+ b0)  -> f16
__global__ __launch_bounds__(128)
void hyper_embed_kernel(const float* __restrict__ hh_new,
                        const float* __restrict__ sx_w0, const float* __restrict__ sx_b0,
                        const float* __restrict__ sh_w0, const float* __restrict__ sh_b0,
                        const float* __restrict__ bh_w0,
                        _Float16* __restrict__ tsx, _Float16* __restrict__ tsh,
                        _Float16* __restrict__ tbh) {
  __shared__ float s_hh[256];
  int b = blockIdx.x, t = threadIdx.x;       // t in [0,128)
  s_hh[t]       = hh_new[(size_t)b * kHH + t];
  s_hh[t + 128] = hh_new[(size_t)b * kHH + 128 + t];
  __syncthreads();
  const float* wx = sx_w0 + (size_t)t * kHH;
  const float* wh = sh_w0 + (size_t)t * kHH;
  const float* wb = bh_w0 + (size_t)t * kHH;
  float ax = 0.f, ah = 0.f, ab = 0.f;
  #pragma unroll 4
  for (int i = 0; i < kHH; ++i) {
    float hv = s_hh[i];
    ax = fmaf(hv, wx[i], ax);
    ah = fmaf(hv, wh[i], ah);
    ab = fmaf(hv, wb[i], ab);
  }
  tsx[(size_t)b * 128 + t] = (_Float16)(ax + sx_b0[t]);
  tsh[(size_t)b * 128 + t] = (_Float16)(ah + sh_b0[t]);
  tbh[(size_t)b * 128 + t] = (_Float16)ab;
}

// ---------------------------------------------------------------- main cell
__global__ __launch_bounds__(256)
void main_cell_kernel(const float* __restrict__ Wx, const float* __restrict__ Wh,
                      const float* __restrict__ scx, const float* __restrict__ sch,
                      const float* __restrict__ sbh,
                      const float* __restrict__ c_total_in,
                      const float* __restrict__ lnh_w, const float* __restrict__ lnh_b,
                      const float* __restrict__ lnc_w, const float* __restrict__ lnc_b,
                      float* __restrict__ out_h, float* __restrict__ out_h_total,
                      float* __restrict__ out_c_total) {
  __shared__ float red[256];
  int b = blockIdx.x, t = threadIdx.x;
  size_t base = (size_t)b * (4 * kH);
  float g[4][4], ln[4][4];
  #pragma unroll
  for (int c = 0; c < 4; ++c)
    #pragma unroll
    for (int j = 0; j < 4; ++j) {
      size_t idx = base + c * kH + j * 256 + t;
      g[c][j] = scx[idx] * Wx[idx] + sch[idx] * Wh[idx] + sbh[idx];
    }
  for (int c = 0; c < 4; ++c) {               // chunk-LN, 4 chunks of 1024
    float ls = g[c][0] + g[c][1] + g[c][2] + g[c][3];
    float mu = block_sum_256(ls, red) * (1.0f / 1024.0f);
    float ls2 = 0.f;
    #pragma unroll
    for (int j = 0; j < 4; ++j) { float d = g[c][j] - mu; ls2 += d * d; }
    float var = block_sum_256(ls2, red) * (1.0f / 1024.0f);
    float rs = rsqrtf(var + kEps);
    #pragma unroll
    for (int j = 0; j < 4; ++j) {
      int n = c * kH + j * 256 + t;
      ln[c][j] = (g[c][j] - mu) * rs * lnh_w[n] + lnh_b[n];
    }
  }
  float c_new[4], lsum = 0.f;
  #pragma unroll
  for (int j = 0; j < 4; ++j) {
    int n = j * 256 + t;
    float c_old = c_total_in[(size_t)b * (kH + kHH) + n];
    float cn = sigmoidf_(ln[1][j] + kForgetBias) * c_old +
               sigmoidf_(ln[0][j]) * tanhf(ln[3][j]);
    c_new[j] = cn;
    lsum += cn;
  }
  float mu = block_sum_256(lsum, red) * (1.0f / 1024.0f);
  float ls2 = 0.f;
  #pragma unroll
  for (int j = 0; j < 4; ++j) { float d = c_new[j] - mu; ls2 += d * d; }
  float var = block_sum_256(ls2, red) * (1.0f / 1024.0f);
  float rs = rsqrtf(var + kEps);
  #pragma unroll
  for (int j = 0; j < 4; ++j) {
    int n = j * 256 + t;
    float lnc = (c_new[j] - mu) * rs * lnc_w[n] + lnc_b[n];
    float h = sigmoidf_(ln[2][j]) * tanhf(lnc);
    out_h[(size_t)b * kH + n] = h;
    out_h_total[(size_t)b * (kH + kHH) + n] = h;
    out_c_total[(size_t)b * (kH + kHH) + n] = c_new[j];
  }
}

// ---------------------------------------------------------------- launch
extern "C" void kernel_launch(void* const* d_in, const int* in_sizes, int n_in,
                              void* d_out, int out_size, void* d_ws, size_t ws_size,
                              hipStream_t stream) {
  (void)in_sizes; (void)n_in; (void)out_size; (void)ws_size;
  const float* x        = (const float*)d_in[0];
  const float* h_total  = (const float*)d_in[1];
  const float* c_total  = (const float*)d_in[2];
  const float* hyp_w_ih = (const float*)d_in[3];
  const float* hyp_w_hh = (const float*)d_in[4];
  const float* hyp_lnh_w= (const float*)d_in[5];
  const float* hyp_lnh_b= (const float*)d_in[6];
  const float* hyp_lnc_w= (const float*)d_in[7];
  const float* hyp_lnc_b= (const float*)d_in[8];
  const float* w_ih     = (const float*)d_in[9];
  const float* w_hh     = (const float*)d_in[10];
  const float* lnh_w    = (const float*)d_in[11];
  const float* lnh_b    = (const float*)d_in[12];
  const float* lnc_w    = (const float*)d_in[13];
  const float* lnc_b    = (const float*)d_in[14];
  const float* sx_w0    = (const float*)d_in[15];
  const float* sx_b0    = (const float*)d_in[16];
  const float* sx_w1    = (const float*)d_in[17];
  const float* sh_w0    = (const float*)d_in[18];
  const float* sh_b0    = (const float*)d_in[19];
  const float* sh_w1    = (const float*)d_in[20];
  const float* bh_w0    = (const float*)d_in[21];
  const float* bh_w1    = (const float*)d_in[22];

  float* out_h    = (float*)d_out;
  float* out_htot = out_h + (size_t)kB * kH;
  float* out_ctot = out_htot + (size_t)kB * (kH + kHH);

  // Bump allocator over d_ws.
  char* wp = (char*)d_ws;
  auto carve = [&](size_t bytes) {
    void* r = (void*)wp;
    wp += (bytes + 255) & ~(size_t)255;
    return r;
  };
  _Float16* x16       = (_Float16*)carve((size_t)kB * kD * 2);
  _Float16* h16       = (_Float16*)carve((size_t)kB * kH * 2);
  _Float16* hh16      = (_Float16*)carve((size_t)kB * kHH * 2);
  _Float16* wih16     = (_Float16*)carve((size_t)4 * kH * kD * 2);
  _Float16* whh16     = (_Float16*)carve((size_t)4 * kH * kH * 2);
  _Float16* hypwih16  = (_Float16*)carve((size_t)4 * kHH * kD * 2);
  _Float16* hypwhh16  = (_Float16*)carve((size_t)4 * kHH * kHH * 2);
  _Float16* sxw1_16   = (_Float16*)carve((size_t)4 * kH * kE * 2);
  _Float16* shw1_16   = (_Float16*)carve((size_t)4 * kH * kE * 2);
  _Float16* bhw1_16   = (_Float16*)carve((size_t)4 * kH * kE * 2);
  _Float16* tsx16     = (_Float16*)carve((size_t)kB * 4 * kE * 2);
  _Float16* tsh16     = (_Float16*)carve((size_t)kB * 4 * kE * 2);
  _Float16* tbh16     = (_Float16*)carve((size_t)kB * 4 * kE * 2);
  float* hyp_wx = (float*)carve((size_t)kB * 4 * kHH * 4);
  float* hyp_wh = (float*)carve((size_t)kB * 4 * kHH * 4);
  float* hh_new = (float*)carve((size_t)kB * kHH * 4);
  float* Wx     = (float*)carve((size_t)kB * 4 * kH * 4);
  float* Wh     = (float*)carve((size_t)kB * 4 * kH * 4);
  float* scx    = (float*)carve((size_t)kB * 4 * kH * 4);
  float* sch    = (float*)carve((size_t)kB * 4 * kH * 4);
  float* sbh    = (float*)carve((size_t)kB * 4 * kH * 4);

  auto cvt = [&](const float* s, _Float16* d, long long n) {
    int blocks = (int)((n + 255) / 256);
    cvt_f32_to_f16<<<blocks, 256, 0, stream>>>(s, d, n);
  };
  cvt(x, x16, (long long)kB * kD);
  cvt(w_ih, wih16, (long long)4 * kH * kD);
  cvt(w_hh, whh16, (long long)4 * kH * kH);
  cvt(hyp_w_ih, hypwih16, (long long)4 * kHH * kD);
  cvt(hyp_w_hh, hypwhh16, (long long)4 * kHH * kHH);
  cvt(sx_w1, sxw1_16, (long long)4 * kH * kE);
  cvt(sh_w1, shw1_16, (long long)4 * kH * kE);
  cvt(bh_w1, bhw1_16, (long long)4 * kH * kE);
  split_h_kernel<<<kB, 256, 0, stream>>>(h_total, h16, hh16);

  // Hyper-LSTM linear layers.  grid = (N/128, M/64), 128 threads.
  wmma_gemm_f16<kD, kD, 4 * kHH, kD>
      <<<dim3((4 * kHH) / 128, kB / 64), 128, 0, stream>>>(x16, hypwih16, hyp_wx);
  wmma_gemm_f16<kHH, kHH, 4 * kHH, kHH>
      <<<dim3((4 * kHH) / 128, kB / 64), 128, 0, stream>>>(hh16, hypwhh16, hyp_wh);
  hyper_cell_kernel<<<kB, 256, 0, stream>>>(hyp_wx, hyp_wh, c_total,
                                            hyp_lnh_w, hyp_lnh_b,
                                            hyp_lnc_w, hyp_lnc_b,
                                            hh_new, out_htot, out_ctot);
  // First-stage hypernet einsums (f32 dot, f16 out).
  hyper_embed_kernel<<<kB, 128, 0, stream>>>(hh_new, sx_w0, sx_b0, sh_w0, sh_b0,
                                             bh_w0, tsx16, tsh16, tbh16);
  // Second-stage einsums as K=32 WMMA GEMMs, one per gate.
  for (int gt = 0; gt < 4; ++gt) {
    wmma_gemm_f16<4 * kE, kE, 4 * kH, kE>
        <<<dim3(kH / 128, kB / 64), 128, 0, stream>>>(
            tsx16 + gt * kE, sxw1_16 + (size_t)gt * kH * kE, scx + gt * kH);
    wmma_gemm_f16<4 * kE, kE, 4 * kH, kE>
        <<<dim3(kH / 128, kB / 64), 128, 0, stream>>>(
            tsh16 + gt * kE, shw1_16 + (size_t)gt * kH * kE, sch + gt * kH);
    wmma_gemm_f16<4 * kE, kE, 4 * kH, kE>
        <<<dim3(kH / 128, kB / 64), 128, 0, stream>>>(
            tbh16 + gt * kE, bhw1_16 + (size_t)gt * kH * kE, sbh + gt * kH);
  }
  // Main GEMMs.
  wmma_gemm_f16<kD, kD, 4 * kH, kD>
      <<<dim3((4 * kH) / 128, kB / 64), 128, 0, stream>>>(x16, wih16, Wx);
  wmma_gemm_f16<kH, kH, 4 * kH, kH>
      <<<dim3((4 * kH) / 128, kB / 64), 128, 0, stream>>>(h16, whh16, Wh);
  // Fused gates + chunk-LN + cell update + cell-LN + outputs.
  main_cell_kernel<<<kB, 256, 0, stream>>>(Wx, Wh, scx, sch, sbh, c_total,
                                           lnh_w, lnh_b, lnc_w, lnc_b,
                                           out_h, out_htot, out_ctot);
}
